// Decoder_35296041239139
// MI455X (gfx1250) — compile-verified, matrix operand order checked by reading, not run on hardware
//
#include <hip/hip_runtime.h>
#include <hip/hip_bf16.h>

#define OBS_LEN 20
#define KK 20
#define BB 4096
#define HH 128
#define MH 64
#define GG 512              // 4*H
#define ROWS (KK * BB)      // 81920
#define NW 4                // waves per block
#define BLOCK (NW * 32)
#define ROWS_PER_BLOCK (NW * 16)

typedef __attribute__((ext_vector_type(16))) __bf16 v16bf;
typedef __attribute__((ext_vector_type(8)))  __bf16 v8bf;
typedef __attribute__((ext_vector_type(4)))  __bf16 v4bf;
typedef __attribute__((ext_vector_type(8)))  float  v8f;

__device__ __forceinline__ v8f wmma_bf16(v16bf a, v16bf b, v8f c) {
  return __builtin_amdgcn_wmma_f32_16x16x32_bf16(false, a, false, b, (short)0, c,
                                                 false, false);
}

__device__ __forceinline__ v8f vzero8() {
  v8f z;
#pragma unroll
  for (int i = 0; i < 8; ++i) z[i] = 0.0f;
  return z;
}

// One 16x32 bf16 operand tile (A and B share per-lane addressing for
// row-major, K-contiguous storage) loaded from LDS as 2x b128.
__device__ __forceinline__ v16bf lds_tile16(const __bf16* base, int strideElems, int k0) {
  const int lm = threadIdx.x & 15;
  const int ho = ((threadIdx.x >> 4) & 1) << 3;   // lanes 16-31 take the +8/+24 K runs
  const __bf16* p = base + lm * strideElems + k0 + ho;
  v8bf lo = *(const v8bf*)(p);
  v8bf hi = *(const v8bf*)(p + 16);
  return __builtin_shufflevector(lo, hi, 0, 1, 2, 3, 4, 5, 6, 7,
                                         8, 9, 10, 11, 12, 13, 14, 15);
}

// A tile straight from global fp32 (row pointer is per-lane), cvt to bf16.
__device__ __forceinline__ v16bf gbl_tile16_f32(const float* rowptr, int k0) {
  const int ho = ((threadIdx.x >> 4) & 1) << 3;
  const float* p = rowptr + k0 + ho;
  v16bf r;
#pragma unroll
  for (int i = 0; i < 8; ++i) r[i] = (__bf16)p[i];
#pragma unroll
  for (int i = 0; i < 8; ++i) r[8 + i] = (__bf16)p[16 + i];
  return r;
}

__device__ __forceinline__ float hsig(float x) {
  return fminf(fmaxf(x * (1.0f / 6.0f) + 0.5f, 0.0f), 1.0f);
}
__device__ __forceinline__ float htanh(float x) {
  return fminf(fmaxf(x, -1.0f), 1.0f);
}

__global__ __launch_bounds__(BLOCK) void decoder_lstm_kernel(
    const float* __restrict__ obs, const float* __restrict__ hidden,
    const float* __restrict__ Wm1, const float* __restrict__ bm1,
    const float* __restrict__ Wm2, const float* __restrict__ bm2,
    const float* __restrict__ Wih, const float* __restrict__ Whh,
    const float* __restrict__ bih, const float* __restrict__ bhh,
    const float* __restrict__ Wout, const float* __restrict__ bout,
    float* __restrict__ out) {
  // 131072 + 6144 + 4096 + 16384 = 157,696 B static LDS
  __shared__ __align__(16) __bf16 s_whh[GG * HH];     // bf16 W_hh; first 32KB overlaid by Wm1/Wm2 in phase 1
  __shared__ float s_wio[3 * GG];                     // [W_ih[:,0] | W_ih[:,1] | b_ih+b_hh]
  __shared__ __align__(16) __bf16 s_wout[16 * HH];    // W_out padded to 16 rows
  __shared__ __align__(16) __bf16 s_h[NW * 16 * HH];  // per-wave h staging (bf16)

  const int tid = threadIdx.x;
  const int wave = tid >> 5;
  const int lm = tid & 15;
  const int half8 = ((tid >> 4) & 1) << 3;            // D-tile row offset for upper lanes
  const int rowBase = (blockIdx.x * NW + wave) * 16;
  __bf16* hw = s_h + wave * (16 * HH);

  // ---------------- Phase 0: stage MLP weights (overlay) + gate tables ----
  {
    const float4* s1 = (const float4*)Wm1;            // 64x128 -> bf16 @ offset 0
    for (int i = tid; i < (MH * HH) / 4; i += BLOCK) {
      float4 v = s1[i];
      v4bf d; d[0] = (__bf16)v.x; d[1] = (__bf16)v.y; d[2] = (__bf16)v.z; d[3] = (__bf16)v.w;
      *(v4bf*)(s_whh + i * 4) = d;
    }
    const float4* s2 = (const float4*)Wm2;            // 128x64 -> bf16 @ offset 8192
    for (int i = tid; i < (HH * MH) / 4; i += BLOCK) {
      float4 v = s2[i];
      v4bf d; d[0] = (__bf16)v.x; d[1] = (__bf16)v.y; d[2] = (__bf16)v.z; d[3] = (__bf16)v.w;
      *(v4bf*)(s_whh + MH * HH + i * 4) = d;
    }
    for (int g = tid; g < GG; g += BLOCK) {
      s_wio[g] = Wih[g * 2 + 0];
      s_wio[GG + g] = Wih[g * 2 + 1];
      s_wio[2 * GG + g] = bih[g] + bhh[g];
    }
    for (int i = tid; i < 16 * HH; i += BLOCK)
      s_wout[i] = (i < 2 * HH) ? (__bf16)Wout[i] : (__bf16)0.0f;
  }
  __syncthreads();

  // ---------------- Phase 1: initial-hidden MLP (per-wave, WMMA) ----------
  {
    const float* Arow = hidden + (size_t)(rowBase + lm) * HH;
    v16bf aG[4];
#pragma unroll
    for (int kc = 0; kc < 4; ++kc) aG[kc] = gbl_tile16_f32(Arow, kc * 32);

    // GEMM1: (16x128)x(128x64) + bias, leaky_relu -> t1 bf16 (stride 64) in hw
#pragma unroll
    for (int jn = 0; jn < 4; ++jn) {
      v8f acc = vzero8();
#pragma unroll
      for (int kc = 0; kc < 4; ++kc)
        acc = wmma_bf16(aG[kc], lds_tile16(s_whh + (jn * 16) * HH, HH, kc * 32), acc);
      const int col = jn * 16 + lm;
      const float b1 = bm1[col];
#pragma unroll
      for (int r = 0; r < 8; ++r) {
        float v = acc[r] + b1;
        v = (v > 0.0f) ? v : 0.01f * v;
        hw[(r + half8) * MH + col] = (__bf16)v;
      }
    }

    // GEMM2: (16x64)x(64x128) + bias -> h0 held in regs, then restage (stride 128)
    v8f h0[8];
#pragma unroll
    for (int jn = 0; jn < 8; ++jn) {
      v8f acc = vzero8();
#pragma unroll
      for (int kc = 0; kc < 2; ++kc)
        acc = wmma_bf16(lds_tile16(hw, MH, kc * 32),
                        lds_tile16(s_whh + MH * HH + (jn * 16) * MH, MH, kc * 32), acc);
      const float b2 = bm2[jn * 16 + lm];
#pragma unroll
      for (int r = 0; r < 8; ++r) acc[r] += b2;
      h0[jn] = acc;
    }
#pragma unroll
    for (int jn = 0; jn < 8; ++jn)
#pragma unroll
      for (int r = 0; r < 8; ++r)
        hw[(r + half8) * HH + jn * 16 + lm] = (__bf16)h0[jn][r];
  }
  __syncthreads();

  // ---------------- Phase 2: stage W_hh (overwrites Wm1/Wm2 overlay) ------
  {
    const float4* s = (const float4*)Whh;             // 512*128/4
    for (int i = tid; i < (GG * HH) / 4; i += BLOCK) {
      float4 v = s[i];
      v4bf d; d[0] = (__bf16)v.x; d[1] = (__bf16)v.y; d[2] = (__bf16)v.z; d[3] = (__bf16)v.w;
      *(v4bf*)(s_whh + i * 4) = d;
    }
  }
  __syncthreads();

  // ---------------- Phase 3: fused 20-step LSTM scan ----------------------
  float cst[8][8];                                    // fp32 cell state in VGPRs
#pragma unroll
  for (int jn = 0; jn < 8; ++jn)
#pragma unroll
    for (int r = 0; r < 8; ++r) cst[jn][r] = 0.0f;

  v16bf a[4];
#pragma unroll
  for (int kc = 0; kc < 4; ++kc) a[kc] = lds_tile16(hw, HH, kc * 32);

  const float bo = bout[lm & 1];
  const int bbase = rowBase & (BB - 1);               // tiles never straddle a k boundary

#pragma unroll 1
  for (int t = 0; t < OBS_LEN; ++t) {
    const int srcIdx = (t == 0) ? 0 : (t - 1);
    const float* xt = obs + (size_t)srcIdx * BB * 2;
    float x0[8], x1[8];
#pragma unroll
    for (int r = 0; r < 8; ++r) {
      float2 xv = *(const float2*)(xt + (size_t)(bbase + r + half8) * 2);
      x0[r] = xv.x; x1[r] = xv.y;
    }

#pragma unroll
    for (int jn = 0; jn < 8; ++jn) {
      v8f gi = vzero8(), gf = vzero8(), gg = vzero8(), go = vzero8();
#pragma unroll
      for (int kc = 0; kc < 4; ++kc) {
        const int k0 = kc * 32;
        gi = wmma_bf16(a[kc], lds_tile16(s_whh + (0 * HH + jn * 16) * HH, HH, k0), gi);
        gf = wmma_bf16(a[kc], lds_tile16(s_whh + (1 * HH + jn * 16) * HH, HH, k0), gf);
        gg = wmma_bf16(a[kc], lds_tile16(s_whh + (2 * HH + jn * 16) * HH, HH, k0), gg);
        go = wmma_bf16(a[kc], lds_tile16(s_whh + (3 * HH + jn * 16) * HH, HH, k0), go);
      }
      const int ci = 0 * HH + jn * 16 + lm;
      const int cf = 1 * HH + jn * 16 + lm;
      const int cg = 2 * HH + jn * 16 + lm;
      const int co = 3 * HH + jn * 16 + lm;
      const float wi0 = s_wio[ci], wi1 = s_wio[GG + ci], bi_ = s_wio[2 * GG + ci];
      const float wf0 = s_wio[cf], wf1 = s_wio[GG + cf], bf_ = s_wio[2 * GG + cf];
      const float wg0 = s_wio[cg], wg1 = s_wio[GG + cg], bg_ = s_wio[2 * GG + cg];
      const float wo0 = s_wio[co], wo1 = s_wio[GG + co], bo_ = s_wio[2 * GG + co];
#pragma unroll
      for (int r = 0; r < 8; ++r) {
        const float giv = gi[r] + x0[r] * wi0 + x1[r] * wi1 + bi_;
        const float gfv = gf[r] + x0[r] * wf0 + x1[r] * wf1 + bf_;
        const float ggv = gg[r] + x0[r] * wg0 + x1[r] * wg1 + bg_;
        const float gov = go[r] + x0[r] * wo0 + x1[r] * wo1 + bo_;
        const float cn = hsig(gfv) * cst[jn][r] + hsig(giv) * htanh(ggv);
        cst[jn][r] = cn;
        const float hn = hsig(gov) * htanh(cn);
        hw[(r + half8) * HH + jn * 16 + lm] = (__bf16)hn;
      }
    }

    // Reload A tiles (new h, feeds next step) + 2-wide output projection.
    v8f oacc = vzero8();
#pragma unroll
    for (int kc = 0; kc < 4; ++kc) {
      a[kc] = lds_tile16(hw, HH, kc * 32);
      oacc = wmma_bf16(a[kc], lds_tile16(s_wout, HH, kc * 32), oacc);
    }
    if (lm < 2) {
      float* op = out + (size_t)t * (ROWS * 2) + (size_t)(rowBase + half8) * 2 + lm;
#pragma unroll
      for (int r = 0; r < 8; ++r) op[r * 2] = oacc[r] + bo;
    }
  }
}

extern "C" void kernel_launch(void* const* d_in, const int* in_sizes, int n_in,
                              void* d_out, int out_size, void* d_ws, size_t ws_size,
                              hipStream_t stream) {
  (void)in_sizes; (void)n_in; (void)out_size; (void)d_ws; (void)ws_size;
  const float* obs = (const float*)d_in[0];
  const float* hid = (const float*)d_in[1];
  const float* Wm1 = (const float*)d_in[2];
  const float* bm1 = (const float*)d_in[3];
  const float* Wm2 = (const float*)d_in[4];
  const float* bm2 = (const float*)d_in[5];
  const float* Wih = (const float*)d_in[6];
  const float* Whh = (const float*)d_in[7];
  const float* bih = (const float*)d_in[8];
  const float* bhh = (const float*)d_in[9];
  const float* Wo  = (const float*)d_in[10];
  const float* bo  = (const float*)d_in[11];
  float* out = (float*)d_out;

  dim3 grid(ROWS / ROWS_PER_BLOCK);   // 1280 blocks
  dim3 block(BLOCK);                  // 128 threads = 4 waves
  hipLaunchKernelGGL(decoder_lstm_kernel, grid, block, 0, stream,
                     obs, hid, Wm1, bm1, Wm2, bm2, Wih, Whh, bih, bhh, Wo, bo, out);
}
